// MOE_26637387169872
// MI455X (gfx1250) — compile-verified, matrix operand order checked by reading, not run on hardware
//
#include <hip/hip_runtime.h>
#include <hip/hip_bf16.h>
#include <stdint.h>

// ---------------------------------------------------------------------------
// MoE for MI455X (gfx1250): bf16 WMMA expert FFNs with async-LDS double
// buffering, fp32 router.  T=8192, D=1024, DFF=4096, 5 FFN experts +
// const/copy/zero experts.
// ---------------------------------------------------------------------------

typedef __attribute__((ext_vector_type(16))) __bf16        v16bf;
typedef __attribute__((ext_vector_type(8)))  float         v8f;
typedef __attribute__((ext_vector_type(4)))  unsigned int  u32x4;
typedef __attribute__((ext_vector_type(4)))  int           i32x4;

union FragBF { v16bf v; u32x4 q[2]; };

typedef __attribute__((address_space(1))) i32x4 gi32x4;
typedef __attribute__((address_space(3))) i32x4 li32x4;

// 16B global -> LDS async copy (ASYNCcnt-tracked, no VGPR round trip).
static __device__ __forceinline__ void async_copy_b128(
    unsigned short* l, const unsigned short* g) {
#if __has_builtin(__builtin_amdgcn_global_load_async_to_lds_b128)
  __builtin_amdgcn_global_load_async_to_lds_b128(
      (gi32x4*)(i32x4*)(void*)const_cast<unsigned short*>(g),
      (li32x4*)(i32x4*)(void*)l, 0, 0);
#else
  unsigned loff =
      (unsigned)(unsigned long long)(__attribute__((address_space(3))) void*)l;
  asm volatile("global_load_async_to_lds_b128 %0, %1, off"
               :: "v"(loff), "v"(g) : "memory");
#endif
}

template <int N>
static __device__ __forceinline__ void wait_async() {
#if __has_builtin(__builtin_amdgcn_s_wait_asynccnt)
  __builtin_amdgcn_s_wait_asynccnt(N);
#else
  asm volatile("s_wait_asynccnt %0" :: "n"(N) : "memory");
#endif
}

static __device__ __forceinline__ unsigned short f32_to_bf16(float f) {
  union { float f; unsigned u; } c; c.f = f;
  unsigned u = c.u;
  unsigned r = u + 0x7FFFu + ((u >> 16) & 1u);   // round-to-nearest-even
  return (unsigned short)(r >> 16);
}

static __device__ __forceinline__ float gelu_tanh(float x) {
  const float k0 = 0.7978845608028654f, k1 = 0.044715f;
  return 0.5f * x * (1.0f + tanhf(k0 * (x + k1 * x * x * x)));
}

// 16x32 bf16 fragment from LDS (A-layout; also valid as B for C = A*W^T with
// row-major W[N,K]).  ISA 7.12.2: lanes 0-15: M=lane, K={0..7,16..23};
// lanes 16-31: K={8..15,24..31}.
static __device__ __forceinline__ v16bf load_frag_lds(
    const unsigned short* lds, int row0, int ldstride, int lane) {
  int r  = row0 + (lane & 15);
  int hi = (lane >> 4) * 8;
  FragBF f;
  f.q[0] = *(const u32x4*)(lds + r * ldstride + hi);
  f.q[1] = *(const u32x4*)(lds + r * ldstride + 16 + hi);
  return f.v;
}

// ---------------------------------------------------------------------------
// Router: wave32 per token.
// ---------------------------------------------------------------------------
__global__ __launch_bounds__(256) void router_kernel(
    const float* __restrict__ x, const float* __restrict__ wg1,
    const float* __restrict__ wg2, float* __restrict__ logits_out,
    float* __restrict__ wgate, int T, int D) {
  int wv   = (int)((blockIdx.x * blockDim.x + threadIdx.x) >> 5);
  int lane = threadIdx.x & 31;
  if (wv >= T) return;
  const float* xr  = x   + (size_t)wv * D;
  const float* w1a = wg1 + (size_t)lane * D;
  const float* w1b = wg1 + (size_t)(lane + 32) * D;
  float h0 = 0.f, h1 = 0.f;
  for (int i = 0; i < D; i += 4) {
    float4 xv = *(const float4*)(xr + i);
    float4 wa = *(const float4*)(w1a + i);
    float4 wb = *(const float4*)(w1b + i);
    h0 += xv.x * wa.x + xv.y * wa.y + xv.z * wa.z + xv.w * wa.w;
    h1 += xv.x * wb.x + xv.y * wb.y + xv.z * wb.z + xv.w * wb.w;
  }
  float t0 = tanhf(h0), t1 = tanhf(h1);
  float lg[8] = {0.f, 0.f, 0.f, 0.f, 0.f, 0.f, 0.f, 0.f};
  for (int j = 0; j < 32; ++j) {
    float a = __shfl(t0, j, 32);
    float b = __shfl(t1, j, 32);
#pragma unroll
    for (int e = 0; e < 8; ++e)
      lg[e] += a * wg2[e * 64 + j] + b * wg2[e * 64 + 32 + j];
  }
  float mx = lg[0];
#pragma unroll
  for (int e = 1; e < 8; ++e) mx = fmaxf(mx, lg[e]);
  float g[8], s = 0.f;
#pragma unroll
  for (int e = 0; e < 8; ++e) { g[e] = expf(lg[e] - mx); s += g[e]; }
#pragma unroll
  for (int e = 0; e < 8; ++e) g[e] /= s;
  int i1 = 0;
#pragma unroll
  for (int e = 1; e < 8; ++e) if (g[e] > g[i1]) i1 = e;
  int i2 = (i1 == 0) ? 1 : 0;
#pragma unroll
  for (int e = 0; e < 8; ++e) if (e != i1 && g[e] > g[i2]) i2 = e;
  float v1 = (i1 == 7) ? 0.f : g[i1];
  float v2 = (i2 == 7) ? 0.f : g[i2];
  float inv = 1.f / (v1 + v2);
  v1 *= inv; v2 *= inv;
  if (lane < 8) {
    logits_out[(size_t)wv * 8 + lane] = lg[lane];
    wgate[(size_t)wv * 8 + lane] = (lane == i1) ? v1 : (lane == i2) ? v2 : 0.f;
  }
}

// ---------------------------------------------------------------------------
// fp32 -> bf16 (optionally scaled) grid-stride conversion.
// ---------------------------------------------------------------------------
__global__ __launch_bounds__(256) void cvt_bf16_kernel(
    const float* __restrict__ src, unsigned short* __restrict__ dst,
    long long n, float scale) {
  long long i = (long long)blockIdx.x * blockDim.x + threadIdx.x;
  long long stride = (long long)gridDim.x * blockDim.x;
  for (; i < n; i += stride) dst[i] = f32_to_bf16(scale * src[i]);
}

// ---------------------------------------------------------------------------
// Constant + Copy expert epilogue; initializes out.
// ---------------------------------------------------------------------------
__global__ __launch_bounds__(256) void init_out_kernel(
    const float* __restrict__ x, const float* __restrict__ const_wg,
    const float* __restrict__ const_c, const float* __restrict__ wgate,
    float* __restrict__ out, int T, int D) {
  int wv   = (int)((blockIdx.x * blockDim.x + threadIdx.x) >> 5);
  int lane = threadIdx.x & 31;
  if (wv >= T) return;
  const float* xr = x + (size_t)wv * D;
  float s0 = 0.f, s1 = 0.f;
  for (int i = lane * 4; i < D; i += 128) {
    float4 xv = *(const float4*)(xr + i);
    float4 c0 = *(const float4*)(const_wg + i);
    float4 c1 = *(const float4*)(const_wg + D + i);
    s0 += 2.f * (xv.x * c0.x + xv.y * c0.y + xv.z * c0.z + xv.w * c0.w);
    s1 += 2.f * (xv.x * c1.x + xv.y * c1.y + xv.z * c1.z + xv.w * c1.w);
  }
#pragma unroll
  for (int off = 16; off > 0; off >>= 1) {
    s0 += __shfl_xor(s0, off, 32);
    s1 += __shfl_xor(s1, off, 32);
  }
  float m  = fmaxf(s0, s1);
  float e0 = expf(s0 - m), e1 = expf(s1 - m);
  float cw0 = e0 / (e0 + e1), cw1 = e1 / (e0 + e1);
  float w5 = wgate[(size_t)wv * 8 + 5];
  float w6 = wgate[(size_t)wv * 8 + 6];
  for (int i = lane * 4; i < D; i += 128) {
    float4 xv = *(const float4*)(xr + i);
    float4 cc = *(const float4*)(const_c + i);
    float4 o;
    o.x = w5 * (cw0 * 2.f * xv.x + cw1 * cc.x) + w6 * 2.f * xv.x;
    o.y = w5 * (cw0 * 2.f * xv.y + cw1 * cc.y) + w6 * 2.f * xv.y;
    o.z = w5 * (cw0 * 2.f * xv.z + cw1 * cc.z) + w6 * 2.f * xv.z;
    o.w = w5 * (cw0 * 2.f * xv.w + cw1 * cc.w) + w6 * 2.f * xv.w;
    *(float4*)(out + (size_t)wv * D + i) = o;
  }
}

// ---------------------------------------------------------------------------
// WMMA bf16 GEMM, C[M,N] = A[M,K] * W[N,K]^T.
// 128x128 block tile, BK=32, double-buffered async-LDS pipeline.
// 8 waves as 4(M) x 2(N); 32x64 per wave -> 8 v_wmma per K-step.
// ---------------------------------------------------------------------------
#define BM 128
#define BN 128
#define BK 32
#define LDP 40   // padded LDS row stride (ushorts): 80B, keeps 16B chunks

#define ISSUE_STAGE(k0, buf)                                                   \
  _Pragma("unroll")                                                            \
  for (int r = 0; r < 2; ++r) {                                                \
    int c  = tid + r * 256;                                                    \
    int rw = c >> 2, cl = (c & 3) * 8;                                         \
    async_copy_b128(lA + (buf) * (BM * LDP) + rw * LDP + cl,                   \
                    A + (size_t)(m0 + rw) * K + (k0) + cl);                    \
    async_copy_b128(lB + (buf) * (BN * LDP) + rw * LDP + cl,                   \
                    Bw + (size_t)(n0 + rw) * K + (k0) + cl);                   \
  }

#define GEMM_BODY(EPILOGUE)                                                    \
  __shared__ unsigned short lA[2 * BM * LDP];                                  \
  __shared__ unsigned short lB[2 * BN * LDP];                                  \
  const int tid   = threadIdx.x;                                               \
  const int lane  = tid & 31;                                                  \
  const int wave  = tid >> 5;                                                  \
  const int waveM = wave & 3;                                                  \
  const int waveN = wave >> 2;                                                 \
  const int m0    = blockIdx.y * BM;                                           \
  const int n0    = blockIdx.x * BN;                                           \
  const v8f vzero = {0.f, 0.f, 0.f, 0.f, 0.f, 0.f, 0.f, 0.f};                  \
  v8f acc[2][4] = {{vzero, vzero, vzero, vzero},                               \
                   {vzero, vzero, vzero, vzero}};                              \
  const int nstage = K / BK;                                                   \
  ISSUE_STAGE(0, 0)                                                            \
  for (int s = 0; s < nstage; ++s) {                                           \
    if (s + 1 < nstage) {                                                      \
      ISSUE_STAGE((s + 1) * BK, (s + 1) & 1)                                   \
      wait_async<4>();   /* stage s complete, stage s+1 in flight */           \
    } else {                                                                   \
      wait_async<0>();                                                         \
    }                                                                          \
    __syncthreads();                                                           \
    const unsigned short* cA = lA + (s & 1) * (BM * LDP);                      \
    const unsigned short* cB = lB + (s & 1) * (BN * LDP);                      \
    v16bf af[2], bfr[4];                                                       \
    _Pragma("unroll")                                                          \
    for (int fm = 0; fm < 2; ++fm)                                             \
      af[fm] = load_frag_lds(cA, waveM * 32 + fm * 16, LDP, lane);             \
    _Pragma("unroll")                                                          \
    for (int fn = 0; fn < 4; ++fn)                                             \
      bfr[fn] = load_frag_lds(cB, waveN * 64 + fn * 16, LDP, lane);            \
    _Pragma("unroll")                                                          \
    for (int fm = 0; fm < 2; ++fm)                                             \
      _Pragma("unroll")                                                        \
      for (int fn = 0; fn < 4; ++fn)                                           \
        acc[fm][fn] = __builtin_amdgcn_wmma_f32_16x16x32_bf16(                 \
            false, af[fm], false, bfr[fn], (short)0, acc[fm][fn],              \
            false, false);                                                     \
    __syncthreads();                                                           \
  }                                                                            \
  const int nloc = lane & 15;                                                  \
  const int mhi  = (lane >> 4) * 8;                                            \
  _Pragma("unroll")                                                            \
  for (int fm = 0; fm < 2; ++fm)                                               \
    _Pragma("unroll")                                                          \
    for (int fn = 0; fn < 4; ++fn)                                             \
      _Pragma("unroll")                                                        \
      for (int j = 0; j < 8; ++j) {                                            \
        int m = m0 + waveM * 32 + fm * 16 + mhi + j;                           \
        int n = n0 + waveN * 64 + fn * 16 + nloc;                              \
        EPILOGUE;                                                              \
      }

// GEMM1: H = bf16(gelu(x2 @ W1^T))
__global__ __launch_bounds__(256) void gemm1_gelu_kernel(
    const unsigned short* __restrict__ A,   // x2 bf16  [M,K]
    const unsigned short* __restrict__ Bw,  // W1 bf16  [N,K]
    unsigned short* __restrict__ H,         // out bf16 [M,N]
    int K, int N) {
  GEMM_BODY(H[(size_t)m * N + n] = f32_to_bf16(gelu_tanh(acc[fm][fn][j])))
}

// GEMM2: out += gate[m,e] * (H @ W2^T)
__global__ __launch_bounds__(256) void gemm2_scale_add_kernel(
    const unsigned short* __restrict__ A,   // H  bf16 [M,K]
    const unsigned short* __restrict__ Bw,  // W2 bf16 [N,K]
    float* __restrict__ out,                // [M,N], accumulated
    const float* __restrict__ wgate, int expert, int K, int N) {
  GEMM_BODY(out[(size_t)m * N + n] +=
                wgate[(size_t)m * 8 + expert] * acc[fm][fn][j])
}

// ---------------------------------------------------------------------------
extern "C" void kernel_launch(void* const* d_in, const int* in_sizes, int n_in,
                              void* d_out, int out_size, void* d_ws,
                              size_t ws_size, hipStream_t stream) {
  const float* x   = (const float*)d_in[0];
  const float* wg1 = (const float*)d_in[1];
  const float* wg2 = (const float*)d_in[2];
  const float* ew1 = (const float*)d_in[3];
  const float* ew2 = (const float*)d_in[4];
  const float* cc  = (const float*)d_in[5];
  const float* cwg = (const float*)d_in[6];

  const int T = 8192, D = 1024, DFF = 4096, NREG = 5;

  float* out    = (float*)d_out;            // [T*D]
  float* logits = out + (size_t)T * D;      // [T*8]

  char* ws = (char*)d_ws;
  float* wgate = (float*)ws;                  ws += (size_t)T * 8 * 4;
  unsigned short* x2b = (unsigned short*)ws;  ws += (size_t)T * D * 2;
  unsigned short* w1b = (unsigned short*)ws;  ws += (size_t)NREG * DFF * D * 2;
  unsigned short* w2b = (unsigned short*)ws;  ws += (size_t)NREG * D * DFF * 2;
  unsigned short* Hb  = (unsigned short*)ws;  ws += (size_t)T * DFF * 2;

  // 1. Router (fp32) -> logits + gate matrix
  router_kernel<<<T / 8, 256, 0, stream>>>(x, wg1, wg2, logits, wgate, T, D);
  // 2. bf16 staging: x*2 and expert weights (weights then live in 192MB L2)
  cvt_bf16_kernel<<<4096, 256, 0, stream>>>(x, x2b, (long long)T * D, 2.0f);
  cvt_bf16_kernel<<<8192, 256, 0, stream>>>(ew1, w1b,
                                            (long long)NREG * DFF * D, 1.0f);
  cvt_bf16_kernel<<<8192, 256, 0, stream>>>(ew2, w2b,
                                            (long long)NREG * D * DFF, 1.0f);
  // 3. Constant + Copy experts initialize out
  init_out_kernel<<<T / 8, 256, 0, stream>>>(x, cwg, cc, wgate, out, T, D);
  // 4. Regular experts: two bf16 WMMA GEMMs each, gate-scaled accumulate
  dim3 g1(DFF / BN, T / BM);
  dim3 g2(D / BN, T / BM);
  for (int e = 0; e < NREG; ++e) {
    gemm1_gelu_kernel<<<g1, 256, 0, stream>>>(
        x2b, w1b + (size_t)e * DFF * D, Hb, D, DFF);
    gemm2_scale_add_kernel<<<g2, 256, 0, stream>>>(
        Hb, w2b + (size_t)e * D * DFF, out, wgate, e, DFF, D);
  }
}
